// NonLocalBlock_80221399155245
// MI455X (gfx1250) — matrix operand order repaired and checked by reading, not compiled, not run on hardware
//
#include <hip/hip_runtime.h>
#include <hip/hip_bf16.h>
#include <cstdint>
#include <math.h>

// ---------------------------------------------------------------------------
// NonLocalBlock for MI455X (gfx1250): flash-attention with f16 WMMA, f32 accum,
// TDM (tensor_load_to_lds) double-buffered K/V staging.
// ---------------------------------------------------------------------------
typedef _Float16 half_t;
typedef __attribute__((ext_vector_type(16))) _Float16     v16h;
typedef __attribute__((ext_vector_type(8)))  float        v8f;
typedef __attribute__((ext_vector_type(4)))  unsigned int u32x4;
typedef __attribute__((ext_vector_type(8)))  int          i32x8;
typedef __attribute__((ext_vector_type(4)))  int          i32x4;

#define BATCH 8
#define CH    256
#define NPIX  4096
#define IDIM  128
#define RED   64
#define PITCH 40    // f16 pitch: 80B rows -> 16B aligned, conflict-free
#define KPITCH 136  // f16 pitch for 128-wide K tile rows (64 dw + 4 dw pad)

// Tensor Data Mover availability (device pass only; host pass parses fallback)
#if defined(__AMDGCN__) && __has_builtin(__builtin_amdgcn_tensor_load_to_lds)
#define USE_TDM 1
#else
#define USE_TDM 0
#endif
#if __has_include(<hip/amd_detail/amd_gfx1250_TDM.h>)
#define TDM_ARGS6 1   // amdgpu-toolchain (clang-23): 6-arg builtin
#else
#define TDM_ARGS6 0   // ROCm 7.2 (clang-22): 5-arg builtin
#endif

union FragU { u32x4 u[2]; v16h h; };

__device__ __forceinline__ v8f wmma16(v16h a, v16h b, v8f c) {
  // D = A(16x32 f16) * B(32x16 f16) + C(16x16 f32)
  return __builtin_amdgcn_wmma_f32_16x16x32_f16(false, a, false, b, (short)0, c,
                                                false, false);
}

// A-fragment (16x32, MxK): lane = row M (l&15); 16 f16 per lane,
// split [half*8 .. +8) and [16+half*8 .. +8) -> two 16B loads
__device__ __forceinline__ v16h load_frag_A(const half_t* base, int pitch) {
  int l = threadIdx.x & 31; int m = l & 15; int hf = l >> 4;
  FragU f;
  f.u[0] = *(const u32x4*)(base + m * pitch + hf * 8);
  f.u[1] = *(const u32x4*)(base + m * pitch + 16 + hf * 8);
  return f.h;
}
// B-fragment (32x16, KxN): lane = col N (l&15); per lane K split
// [half*8 .. +8) and [16+half*8 .. +8) (memory [n][k]) -> two 16B loads
__device__ __forceinline__ v16h load_frag_B(const half_t* base, int pitch) {
  int l = threadIdx.x & 31; int n = l & 15; int hf = l >> 4;
  FragU f;
  f.u[0] = *(const u32x4*)(base + n * pitch + hf * 8);
  f.u[1] = *(const u32x4*)(base + n * pitch + 16 + hf * 8);
  return f.h;
}

#if USE_TDM
// Issue a 2D TDM tile load (f16 elements) global -> LDS.
// D# per CDNA5 ISA ch.8: group0 = {count/lds_addr/global_addr/type},
// group1 = {flags, dims, tile dims, stride}; groups 2/3 unused (2D tensor).
__device__ __forceinline__ void tdm_load_2d(unsigned lds_addr, const void* gaddr,
                                            unsigned flags,
                                            unsigned tensor_d0, unsigned tensor_d1,
                                            unsigned tile_d0, unsigned tile_d1,
                                            unsigned stride0) {
  unsigned long long ga = (unsigned long long)(uintptr_t)gaddr;
  u32x4 g0;
  g0[0] = 1u;                                            // count=1, user mode
  g0[1] = lds_addr;                                      // LDS byte address
  g0[2] = (unsigned)(ga & 0xffffffffu);                  // global_addr[31:0]
  g0[3] = (unsigned)((ga >> 32) & 0x01ffffffu)           // global_addr[56:32]
          | 0x80000000u;                                 // type=2 ("image")
  i32x8 g1;
  g1[0] = (int)flags;                                    // mask=0|data_size|pad
  g1[1] = (int)((tensor_d0 & 0xffffu) << 16);            // dim0[15:0] @ [63:48]
  g1[2] = (int)((tensor_d0 >> 16) | ((tensor_d1 & 0xffffu) << 16));
  g1[3] = (int)((tensor_d1 >> 16) | (tile_d0 << 16));    // tile_dim0 @ [127:112]
  g1[4] = (int)(tile_d1 & 0xffffu);                      // tile_dim1; tile_dim2=0
  g1[5] = (int)stride0;                                  // dim0_stride[31:0]
  g1[6] = 0;                                             // stride hi / dim1_stride
  g1[7] = 0;
  i32x4 z4 = {0, 0, 0, 0};
#if TDM_ARGS6
  i32x8 z8 = {0, 0, 0, 0, 0, 0, 0, 0};
  __builtin_amdgcn_tensor_load_to_lds(g0, g1, z4, z4, z8, 0);
#else
  __builtin_amdgcn_tensor_load_to_lds(g0, g1, z4, z4, 0);
#endif
}
#endif // USE_TDM

// ---------------------------------------------------------------------------
// 1) weights f32 -> f16  (theta | phi | g | out_w), each 32768 elements
// ---------------------------------------------------------------------------
__global__ __launch_bounds__(256) void k_convert_w(const float* th, const float* ph,
                                                   const float* g, const float* ow,
                                                   half_t* dst) {
  int idx = blockIdx.x * 256 + threadIdx.x;      // 0..131071
  int which = idx >> 15; int r = idx & 32767;
  const float* s = (which == 0) ? th : (which == 1) ? ph : (which == 2) ? g : ow;
  dst[idx] = (half_t)s[r];
}

// ---------------------------------------------------------------------------
// 2) QKV projection: per (n-tile of 128, batch, weight) GEMM [128x256]x[256x128]
//    Q,K stored [B][N][I];  V stored transposed [B][I][N] (for P*V B-frags)
// ---------------------------------------------------------------------------
__global__ __launch_bounds__(256) void k_proj(const float* __restrict__ x,
                                              const half_t* __restrict__ wf16,
                                              half_t* Qo, half_t* Ko, half_t* VTo) {
  __shared__ half_t xs[128 * PITCH];    // x tile transposed: [n][c_local] f16
  __shared__ half_t wsb[128 * PITCH];   // weight slab: [i][c_local] f16
  __shared__ half_t tb[128 * 132];      // output transpose staging
  int n0 = blockIdx.x * 128; int b = blockIdx.y; int widx = blockIdx.z;
  int t = threadIdx.x; int w = t >> 5; int l = t & 31; int hf = l >> 4; int m = l & 15;
  const half_t* wsrc = wf16 + (size_t)widx * 32768;
  const float* xb = x + (size_t)b * CH * NPIX + n0;

  v8f acc[8];
  v8f zz = {0,0,0,0,0,0,0,0};
#pragma unroll
  for (int i = 0; i < 8; ++i) acc[i] = zz;

  for (int ks = 0; ks < 8; ++ks) {
    int kc = ks * 32;
    __syncthreads();
    { // stage x tile (f32 [c][n] -> f16 [n][c]) : coalesced float4 reads
      int cl4 = t >> 5; int n4 = t & 31;
#pragma unroll
      for (int cc = 0; cc < 4; ++cc) {
        int c = cl4 * 4 + cc;
        float4 v = *(const float4*)(xb + (size_t)(kc + c) * NPIX + n4 * 4);
        xs[(n4 * 4 + 0) * PITCH + c] = (half_t)v.x;
        xs[(n4 * 4 + 1) * PITCH + c] = (half_t)v.y;
        xs[(n4 * 4 + 2) * PITCH + c] = (half_t)v.z;
        xs[(n4 * 4 + 3) * PITCH + c] = (half_t)v.w;
      }
    }
    { // stage weight slab [128 i][32 c] as u32 copies
      const uint32_t* src = (const uint32_t*)wsrc + (kc >> 1);
      uint32_t* dst = (uint32_t*)wsb;
#pragma unroll
      for (int p = 0; p < 8; ++p) {
        int flat = t + p * 256;
        int i = flat >> 4, cu = flat & 15;
        dst[i * (PITCH / 2) + cu] = src[(size_t)i * 128 + cu];
      }
    }
    __syncthreads();
    v16h a = load_frag_A(xs + w * 16 * PITCH, PITCH);
#pragma unroll
    for (int tt = 0; tt < 8; ++tt) {
      v16h bb = load_frag_B(wsb + tt * 16 * PITCH, PITCH);
      acc[tt] = wmma16(a, bb, acc[tt]);
    }
  }
  __syncthreads();
  // C/D layout: lane n = l&15 (col), VGPR r -> row r + hf*8
  if (widx < 2) { // Q,K: tb[n][i]
#pragma unroll
    for (int tt = 0; tt < 8; ++tt)
#pragma unroll
      for (int r = 0; r < 8; ++r)
        tb[(w * 16 + r + hf * 8) * 132 + tt * 16 + m] = (half_t)acc[tt][r];
  } else {        // V: tb[i][n]  (transpose)
#pragma unroll
    for (int tt = 0; tt < 8; ++tt)
#pragma unroll
      for (int r = 0; r < 8; ++r)
        tb[(tt * 16 + m) * 132 + w * 16 + r + hf * 8] = (half_t)acc[tt][r];
  }
  __syncthreads();
  if (widx < 2) {
    half_t* dsth = (widx == 0 ? Qo : Ko) + ((size_t)b * NPIX + n0) * IDIM;
    uint32_t* d32 = (uint32_t*)dsth;
    for (int p = 0; p < 32; ++p) {
      int flat = p * 256 + t;
      int row = flat >> 6, cu = flat & 63;
      d32[flat] = *(const uint32_t*)&tb[row * 132 + cu * 2];
    }
  } else {
    for (int p = 0; p < 32; ++p) {
      int flat = p * 256 + t;
      int i = flat >> 6, cu = flat & 63;
      uint32_t* d32 = (uint32_t*)(VTo + (size_t)b * IDIM * NPIX + (size_t)i * NPIX + n0);
      d32[cu] = *(const uint32_t*)&tb[i * 132 + cu * 2];
    }
  }
}

// ---------------------------------------------------------------------------
// 3) Flash attention: 128 q-rows per block (16 per wave), stream KV in 32-col
//    tiles. K/V tiles staged by the Tensor Data Mover into double-buffered
//    LDS (TENSORcnt + barrier pipelining); online softmax; O -> [B][N][I] f16
// ---------------------------------------------------------------------------
__global__ __launch_bounds__(256) void k_flash(const half_t* __restrict__ Q,
                                               const half_t* __restrict__ K,
                                               const half_t* __restrict__ VT,
                                               half_t* __restrict__ O) {
  __shared__ half_t pbuf[8 * 16 * PITCH];     // per-wave P staging (A layout)
#if USE_TDM
  __shared__ half_t kbuf[2 * 32 * KPITCH];    // K tile [32 kv][128 i] (+pad)
  __shared__ half_t vbuf[2 * 128 * PITCH];    // V tile [128 i][32 kv] (+pad)
  // group1 flags: data_size=2B (1<<16), pad_enable (1<<20),
  // pad_interval code (bits 24:22), pad_amount code = 3 -> 4 dwords (bits 31:25)
  const unsigned KFLAGS = (1u << 16) | (1u << 20) | (5u << 22) | (3u << 25); // 64dw+4
  const unsigned VFLAGS = (1u << 16) | (1u << 20) | (3u << 22) | (3u << 25); // 16dw+4
#endif
  int b = blockIdx.y; int n0 = blockIdx.x * 128;
  int t = threadIdx.x; int w = t >> 5; int l = t & 31; int hf = l >> 4; int m = l & 15;
  const float KS = 0.12754247f;               // (1/sqrt(128)) * log2(e)

  const half_t* qrow = Q + ((size_t)b * NPIX + n0 + w * 16) * IDIM;
  v16h qf[4];
#pragma unroll
  for (int ks = 0; ks < 4; ++ks) qf[ks] = load_frag_A(qrow + ks * 32, IDIM);

  v8f o[8];
  v8f zz = {0,0,0,0,0,0,0,0};
#pragma unroll
  for (int i = 0; i < 8; ++i) o[i] = zz;
  float mrun[8], lrun[8];
#pragma unroll
  for (int r = 0; r < 8; ++r) { mrun[r] = -1e30f; lrun[r] = 0.f; }

  half_t* pb = pbuf + w * 16 * PITCH;
  const half_t* kbase = K + (size_t)b * NPIX * IDIM;
  const half_t* vbase = VT + (size_t)b * IDIM * NPIX;

#if USE_TDM
  if (w == 0) {  // prologue: TDM tile 0 into buffer 0
    tdm_load_2d((unsigned)(uintptr_t)(void*)kbuf, kbase, KFLAGS,
                128, 32, 128, 32, 128);
    tdm_load_2d((unsigned)(uintptr_t)(void*)vbuf, vbase, VFLAGS,
                32, 128, 32, 128, NPIX);
    __builtin_amdgcn_s_wait_tensorcnt(0);
  }
  __syncthreads();
#endif

  for (int kv0 = 0; kv0 < NPIX; kv0 += 32) {
#if USE_TDM
    int cur = (kv0 >> 5) & 1;
    const half_t* kb = kbuf + cur * (32 * KPITCH);
    const half_t* vb = vbuf + cur * (128 * PITCH);
    if (w == 0 && kv0 + 32 < NPIX) {  // async prefetch of next tile via TDM
      int nxt = cur ^ 1;
      tdm_load_2d((unsigned)(uintptr_t)(void*)(kbuf + nxt * (32 * KPITCH)),
                  kbase + (size_t)(kv0 + 32) * IDIM, KFLAGS, 128, 32, 128, 32, 128);
      tdm_load_2d((unsigned)(uintptr_t)(void*)(vbuf + nxt * (128 * PITCH)),
                  vbase + (kv0 + 32), VFLAGS, 32, 128, 32, 128, NPIX);
    }
#else
    const half_t* kp = kbase + (size_t)kv0 * IDIM;
#endif
    v8f s0 = zz, s1 = zz;
#pragma unroll
    for (int ks = 0; ks < 4; ++ks) {           // S = Q * K^T
#if USE_TDM
      v16h b0 = load_frag_B(kb + ks * 32, KPITCH);
      v16h b1 = load_frag_B(kb + 16 * KPITCH + ks * 32, KPITCH);
#else
      v16h b0 = load_frag_B(kp + ks * 32, IDIM);
      v16h b1 = load_frag_B(kp + 16 * IDIM + ks * 32, IDIM);
#endif
      s0 = wmma16(qf[ks], b0, s0);
      s1 = wmma16(qf[ks], b1, s1);
    }
    float cf[8];
#pragma unroll
    for (int r = 0; r < 8; ++r) {              // online softmax, rows r + hf*8
      float mx = fmaxf(s0[r], s1[r]);
      mx = fmaxf(mx, __shfl_xor(mx, 1));
      mx = fmaxf(mx, __shfl_xor(mx, 2));
      mx = fmaxf(mx, __shfl_xor(mx, 4));
      mx = fmaxf(mx, __shfl_xor(mx, 8));       // 16-lane (row) reduction
      float mnew = fmaxf(mrun[r], mx);
      cf[r] = exp2f((mrun[r] - mnew) * KS);
      float p0 = exp2f((s0[r] - mnew) * KS);
      float p1 = exp2f((s1[r] - mnew) * KS);
      pb[(r + hf * 8) * PITCH + m]      = (half_t)p0;
      pb[(r + hf * 8) * PITCH + 16 + m] = (half_t)p1;
      float ps = p0 + p1;
      ps += __shfl_xor(ps, 1); ps += __shfl_xor(ps, 2);
      ps += __shfl_xor(ps, 4); ps += __shfl_xor(ps, 8);
      lrun[r] = lrun[r] * cf[r] + ps;
      mrun[r] = mnew;
    }
#pragma unroll
    for (int tt = 0; tt < 8; ++tt)
#pragma unroll
      for (int r = 0; r < 8; ++r) o[tt][r] *= cf[r];

    v16h pf = load_frag_A(pb, PITCH);          // P as A-fragment (K=32)
#pragma unroll
    for (int tt = 0; tt < 8; ++tt) {           // O += P * V
#if USE_TDM
      v16h bv = load_frag_B(vb + tt * 16 * PITCH, PITCH);
#else
      v16h bv = load_frag_B(vbase + (size_t)(tt * 16) * NPIX + kv0, NPIX);
#endif
      o[tt] = wmma16(pf, bv, o[tt]);
    }
#if USE_TDM
    if (w == 0) __builtin_amdgcn_s_wait_tensorcnt(0);
    __syncthreads();                           // next buffer ready for all waves
#else
    if (kv0 + 32 < NPIX) {                     // global_prefetch_b8 hints
      __builtin_prefetch(kp + 32 * IDIM, 0, 1);
      __builtin_prefetch(vbase + kv0 + 32, 0, 1);
    }
#endif
  }
  float inv[8];
#pragma unroll
  for (int r = 0; r < 8; ++r) inv[r] = 1.f / lrun[r];
  half_t* orow = O + ((size_t)b * NPIX + n0 + w * 16) * IDIM;
#pragma unroll
  for (int tt = 0; tt < 8; ++tt)
#pragma unroll
    for (int r = 0; r < 8; ++r)
      orow[(size_t)(r + hf * 8) * IDIM + tt * 16 + m] = (half_t)(o[tt][r] * inv[r]);
}

// ---------------------------------------------------------------------------
// 4) Output projection: [128n x 128i] * out_w^T -> [128n x 256c],
//    LDS transpose -> coalesced f32 stores to attn[B][C][N]
// ---------------------------------------------------------------------------
__global__ __launch_bounds__(256) void k_outproj(const half_t* __restrict__ O,
                                                 const half_t* __restrict__ OW,
                                                 float* __restrict__ attn) {
  __shared__ float tb[8 * 64 * 17];     // per-wave [64c][16n] transpose chunk
  int b = blockIdx.y; int n0 = blockIdx.x * 128;
  int t = threadIdx.x; int w = t >> 5; int l = t & 31; int hf = l >> 4; int m = l & 15;
  const half_t* orow = O + ((size_t)b * NPIX + n0 + w * 16) * IDIM;
  v16h af[4];
#pragma unroll
  for (int ks = 0; ks < 4; ++ks) af[ks] = load_frag_A(orow + ks * 32, IDIM);
  float* tbw = tb + w * 64 * 17;
  v8f zz = {0,0,0,0,0,0,0,0};
#pragma unroll
  for (int p = 0; p < 2; ++p) {         // 2 halves of the 256 output channels
    v8f acc[8];
#pragma unroll
    for (int i = 0; i < 8; ++i) acc[i] = zz;
#pragma unroll
    for (int ks = 0; ks < 4; ++ks)
#pragma unroll
      for (int tt = 0; tt < 8; ++tt) {
        v16h bf = load_frag_B(OW + (size_t)(p * 128 + tt * 16) * IDIM + ks * 32, IDIM);
        acc[tt] = wmma16(af[ks], bf, acc[tt]);
      }
#pragma unroll
    for (int g = 0; g < 2; ++g) {       // transpose 64 channels at a time via LDS
#pragma unroll
      for (int tt = 0; tt < 4; ++tt)
#pragma unroll
        for (int r = 0; r < 8; ++r)
          tbw[(tt * 16 + m) * 17 + r + hf * 8] = acc[g * 4 + tt][r];
      // in-wave LDS ordering guarantees write->read
      for (int j = 0; j < 32; ++j) {
        int cl = hf + j * 2;
        float v = tbw[cl * 17 + m];
        int cg = p * 128 + g * 64 + cl;
        attn[((size_t)b * CH + cg) * NPIX + n0 + w * 16 + m] = v;
      }
    }
  }
}

// ---------------------------------------------------------------------------
// 5) BatchNorm stats per channel (biased var, eps=1e-5)
// ---------------------------------------------------------------------------
__global__ __launch_bounds__(256) void k_bn_stats(const float* __restrict__ attn,
                                                  float* meanp, float* rstdp) {
  int c = blockIdx.x; int t = threadIdx.x;
  float s = 0.f, s2 = 0.f;
  for (int b = 0; b < BATCH; ++b) {
    const float4* p = (const float4*)(attn + ((size_t)b * CH + c) * NPIX);
    for (int j = 0; j < 4; ++j) {
      float4 v = p[t + j * 256];
      s  += v.x + v.y + v.z + v.w;
      s2 += v.x * v.x + v.y * v.y + v.z * v.z + v.w * v.w;
    }
  }
  __shared__ float sb[256], sb2[256];
  sb[t] = s; sb2[t] = s2; __syncthreads();
  for (int off = 128; off > 0; off >>= 1) {
    if (t < off) { sb[t] += sb[t + off]; sb2[t] += sb2[t + off]; }
    __syncthreads();
  }
  if (t == 0) {
    float mn = sb[0] / (float)(BATCH * NPIX);
    float vr = sb2[0] / (float)(BATCH * NPIX) - mn * mn;
    meanp[c] = mn; rstdp[c] = rsqrtf(vr + 1e-5f);
  }
}

// ---------------------------------------------------------------------------
// 6) SE: spatial pool, then tiny 2-layer MLP (relu + sigmoid)
// ---------------------------------------------------------------------------
__global__ __launch_bounds__(256) void k_pool(const float* __restrict__ x, float* pooled) {
  int c = blockIdx.x, b = blockIdx.y, t = threadIdx.x;
  const float4* p = (const float4*)(x + ((size_t)b * CH + c) * NPIX);
  float s = 0.f;
  for (int j = 0; j < 4; ++j) { float4 v = p[t + j * 256]; s += v.x + v.y + v.z + v.w; }
  __shared__ float sb[256];
  sb[t] = s; __syncthreads();
  for (int off = 128; off > 0; off >>= 1) { if (t < off) sb[t] += sb[t + off]; __syncthreads(); }
  if (t == 0) pooled[b * CH + c] = sb[0] / (float)NPIX;
}

__global__ __launch_bounds__(256) void k_se(const float* pooled, const float* fc1w,
                                            const float* fc1b, const float* fc2w,
                                            const float* fc2b, float* chw) {
  __shared__ float pl[BATCH * CH];
  __shared__ float hs[BATCH * RED];
  int t = threadIdx.x;
  for (int i = t; i < BATCH * CH; i += 256) pl[i] = pooled[i];
  __syncthreads();
  for (int o = t; o < BATCH * RED; o += 256) {
    int b = o >> 6, r = o & 63; float s = fc1b[r];
    for (int c = 0; c < CH; ++c) s += pl[b * CH + c] * fc1w[r * CH + c];
    hs[o] = fmaxf(s, 0.f);
  }
  __syncthreads();
  for (int o = t; o < BATCH * CH; o += 256) {
    int b = o >> 8, c = o & 255; float s = fc2b[c];
    for (int r = 0; r < RED; ++r) s += hs[b * RED + r] * fc2w[c * RED + r];
    chw[o] = 1.f / (1.f + expf(-s));
  }
}

// ---------------------------------------------------------------------------
// 7) finalize: out = x + bn(attn) * ch_w      (float4 streaming)
// ---------------------------------------------------------------------------
__global__ __launch_bounds__(256) void k_final(const float* __restrict__ x,
                                               const float* __restrict__ attn,
                                               const float* meanp, const float* rstdp,
                                               const float* gamma, const float* beta,
                                               const float* chw, float* __restrict__ out) {
  int f4 = blockIdx.x * 256 + threadIdx.x;  // 2,097,152 float4 groups
  int row = f4 >> 10; int c = row & 255; int b = row >> 8;
  float sc = rstdp[c] * gamma[c];
  float sh = beta[c] - meanp[c] * sc;
  float wv = chw[b * CH + c];
  float4 xv = ((const float4*)x)[f4];
  float4 av = ((const float4*)attn)[f4];
  float4 ov;
  ov.x = xv.x + (av.x * sc + sh) * wv;
  ov.y = xv.y + (av.y * sc + sh) * wv;
  ov.z = xv.z + (av.z * sc + sh) * wv;
  ov.w = xv.w + (av.w * sc + sh) * wv;
  ((float4*)out)[f4] = ov;
}

// ---------------------------------------------------------------------------
extern "C" void kernel_launch(void* const* d_in, const int* in_sizes, int n_in,
                              void* d_out, int out_size, void* d_ws, size_t ws_size,
                              hipStream_t stream) {
  const float* x     = (const float*)d_in[0];
  const float* thw   = (const float*)d_in[1];
  const float* phw   = (const float*)d_in[2];
  const float* gw    = (const float*)d_in[3];
  const float* oww   = (const float*)d_in[4];
  const float* gamma = (const float*)d_in[5];
  const float* beta  = (const float*)d_in[6];
  const float* fc1w  = (const float*)d_in[7];
  const float* fc1b  = (const float*)d_in[8];
  const float* fc2w  = (const float*)d_in[9];
  const float* fc2b  = (const float*)d_in[10];
  float* out = (float*)d_out;

  // workspace layout (~67.4 MB)
  char* ws = (char*)d_ws;
  half_t* wf16 = (half_t*)ws;                                   // 262144 B
  half_t* Q    = (half_t*)(ws + (256 << 10));                   // 8 MB each
  half_t* K    = Q  + (size_t)BATCH * NPIX * IDIM;
  half_t* VT   = K  + (size_t)BATCH * NPIX * IDIM;
  half_t* O    = VT + (size_t)BATCH * NPIX * IDIM;
  float*  attn = (float*)(O + (size_t)BATCH * NPIX * IDIM);     // 32 MB f32
  float*  pooled = attn + (size_t)BATCH * CH * NPIX;            // 8 KB
  float*  chw    = pooled + BATCH * CH;                         // 8 KB
  float*  meanp  = chw + BATCH * CH;                            // 1 KB
  float*  rstdp  = meanp + CH;                                  // 1 KB

  k_convert_w<<<512, 256, 0, stream>>>(thw, phw, gw, oww, wf16);
  k_proj    <<<dim3(32, 8, 3), 256, 0, stream>>>(x, wf16, Q, K, VT);
  k_flash   <<<dim3(32, 8),    256, 0, stream>>>(Q, K, VT, O);
  k_outproj <<<dim3(32, 8),    256, 0, stream>>>(O, wf16 + 3 * 32768, attn);
  k_bn_stats<<<256,            256, 0, stream>>>(attn, meanp, rstdp);
  k_pool    <<<dim3(256, 8),   256, 0, stream>>>(x, pooled);
  k_se      <<<1,              256, 0, stream>>>(pooled, fc1w, fc1b, fc2w, fc2b, chw);
  k_final   <<<8192,           256, 0, stream>>>(x, attn, meanp, rstdp, gamma, beta, chw, out);
}
